// LCALSTM_after_30348238913613
// MI455X (gfx1250) — compile-verified, matrix operand order checked by reading, not run on hardware
//
#include <hip/hip_runtime.h>
#include <hip/hip_bf16.h>
#include <stdint.h>

#define B_SZ  16384
#define DIN   257
#define H_SZ  512
#define DECSZ 512
#define A_SZ  64
#define KPACK 800     // 257 + 512 padded up to a multiple of 32
#define NPACK 2048    // 4H useful output columns of GEMM1 (row 3H skipped)
#define NAC   128     // actor(64)+critic(1) padded to a 128-col tile

// ---- CDNA5 feature gates (device pass only; host pass sees 0 and skips) ----
#if defined(__HIP_DEVICE_COMPILE__) && __has_builtin(__builtin_amdgcn_global_load_async_to_lds_b128)
#define USE_ASYNC_LDS 1
#else
#define USE_ASYNC_LDS 0
#endif
#if defined(__HIP_DEVICE_COMPILE__) && __has_builtin(__builtin_amdgcn_tensor_load_to_lds)
#define USE_TDM 1
#else
#define USE_TDM 0
#endif

typedef __attribute__((ext_vector_type(16))) __bf16 v16bf;
typedef __attribute__((ext_vector_type(8)))  float  v8f;
typedef int v4i __attribute__((vector_size(16)));   // pointee type of async-LDS builtins

__device__ __forceinline__ uint16_t f32_to_bf16_rne(float f) {
    uint32_t u = __float_as_uint(f);
    uint32_t r = u + 0x7FFFu + ((u >> 16) & 1u);
    return (uint16_t)(r >> 16);
}
__device__ __forceinline__ float sigmoidf_(float x) { return 1.0f / (1.0f + __expf(-x)); }

__device__ __forceinline__ void wait_async0() {
#if defined(__HIP_DEVICE_COMPILE__)
#if __has_builtin(__builtin_amdgcn_s_wait_asynccnt)
    __builtin_amdgcn_s_wait_asynccnt(0);
#else
    asm volatile("s_wait_asynccnt 0x0" ::: "memory");
#endif
#endif
}
__device__ __forceinline__ void wait_tensor0() {
#if defined(__HIP_DEVICE_COMPILE__)
#if __has_builtin(__builtin_amdgcn_s_wait_tensorcnt)
    __builtin_amdgcn_s_wait_tensorcnt(0);
#else
    asm volatile("s_wait_tensorcnt 0x0" ::: "memory");
#endif
#endif
}

#if USE_TDM
typedef __attribute__((ext_vector_type(4))) unsigned int u32x4;
typedef __attribute__((ext_vector_type(8))) int          i32x8;
typedef __attribute__((ext_vector_type(4))) int          i32x4;

// TDM: load a [rows=128] x [32 halves] tile from a row-major bf16 matrix with
// row stride K (elements) into LDS with a 40-half padded row stride
// (pad_interval = 16 DWORDs, pad_amount = 4 DWORDs).
__device__ __forceinline__ void tdm_load_tile_2d(const void* gptr, unsigned lds_off,
                                                 int K, int N) {
    unsigned long long ga = (unsigned long long)(uintptr_t)gptr;
    u32x4 g0;
    g0[0] = 1u;                                   // count=1, user descriptor
    g0[1] = lds_off;                              // lds_addr (bytes)
    g0[2] = (unsigned)(ga & 0xFFFFFFFFu);         // global_addr[31:0]
    g0[3] = (unsigned)((ga >> 32) & 0x01FFFFFFu)  // global_addr[56:32]
          | 0x80000000u;                          // type = 2 ("image")
    i32x8 g1;
    g1[0] = (1 << 16)                             // data_size = 2 bytes
          | (1 << 20)                             // pad_enable
          | (3 << 22)                             // pad_interval: 16 DWORDs (32 halves)
          | (3 << 25);                            // pad_amount: 4 DWORDs (8 halves)
    g1[1] = (K & 0xFFFF) << 16;                   // tensor_dim0[15:0]
    g1[2] = ((unsigned)K >> 16) | ((N & 0xFFFF) << 16);   // tensor_dim0 hi | tensor_dim1 lo
    g1[3] = (((unsigned)N >> 16) & 0xFFFF) | (32 << 16);  // tensor_dim1 hi | tile_dim0 = 32
    g1[4] = 128;                                  // tile_dim1 = 128 rows, tile_dim2 = 0
    g1[5] = K;                                    // tensor_dim0_stride[31:0] (elements)
    g1[6] = 0;                                    // stride hi | tensor_dim1_stride lo
    g1[7] = 0;
    i32x4 g2 = {0, 0, 0, 0};
    i32x4 g3 = {0, 0, 0, 0};
#if __clang_major__ >= 23
    i32x8 g4 = {0, 0, 0, 0, 0, 0, 0, 0};
    __builtin_amdgcn_tensor_load_to_lds(g0, g1, g2, g3, g4, 0);
#else
    __builtin_amdgcn_tensor_load_to_lds(g0, g1, g2, g3, 0);
#endif
}
#endif // USE_TDM

// ---------------------------------------------------------------- pack kernels
__global__ __launch_bounds__(256)
void pack_act_kernel(const float* __restrict__ x, const float* __restrict__ h,
                     uint16_t* __restrict__ out) {
    int b = blockIdx.x;
    const float* xr = x + (size_t)b * DIN;
    const float* hr = h + (size_t)b * H_SZ;
    uint16_t* o = out + (size_t)b * KPACK;
    for (int j = threadIdx.x; j < KPACK; j += 256) {
        float v = 0.f;
        if (j < DIN)             v = xr[j];
        else if (j < DIN + H_SZ) v = hr[j - DIN];
        o[j] = f32_to_bf16_rne(v);
    }
}

__global__ __launch_bounds__(256)
void pack_w1_kernel(const float* __restrict__ Wi, const float* __restrict__ bi,
                    const float* __restrict__ Wh, const float* __restrict__ bh,
                    uint16_t* __restrict__ Wp, float* __restrict__ bias) {
    int jp = blockIdx.x;                       // 0..2047 packed row
    int j  = jp + (jp >= 3 * H_SZ ? 1 : 0);    // skip original row 3H (=1536)
    const float* wi = Wi + (size_t)j * DIN;
    const float* wh = Wh + (size_t)j * H_SZ;
    uint16_t* o = Wp + (size_t)jp * KPACK;
    for (int k = threadIdx.x; k < KPACK; k += 256) {
        float v = 0.f;
        if (k < DIN)             v = wi[k];
        else if (k < DIN + H_SZ) v = wh[k - DIN];
        o[k] = f32_to_bf16_rne(v);
    }
    if (threadIdx.x == 0) bias[jp] = bi[j] + bh[j];
}

__global__ __launch_bounds__(256)
void cvt_bf16_kernel(const float* __restrict__ in, uint16_t* __restrict__ out, int n) {
    int i = blockIdx.x * 256 + threadIdx.x;
    if (i < n) out[i] = f32_to_bf16_rne(in[i]);
}

__global__ __launch_bounds__(256)
void pack_wac_kernel(const float* __restrict__ Wa, const float* __restrict__ ba,
                     const float* __restrict__ Wc, const float* __restrict__ bc,
                     uint16_t* __restrict__ Wp, float* __restrict__ bias) {
    int n = blockIdx.x; // 0..127: rows 0..63 actor, 64 critic, 65..127 zero pad
    uint16_t* o = Wp + (size_t)n * DECSZ;
    if (n < A_SZ) {
        const float* src = Wa + (size_t)n * DECSZ;
        for (int k = threadIdx.x; k < DECSZ; k += 256) o[k] = f32_to_bf16_rne(src[k]);
        if (threadIdx.x == 0) bias[n] = ba[n];
    } else if (n == A_SZ) {
        for (int k = threadIdx.x; k < DECSZ; k += 256) o[k] = f32_to_bf16_rne(Wc[k]);
        if (threadIdx.x == 0) bias[n] = bc[0];
    } else {
        for (int k = threadIdx.x; k < DECSZ; k += 256) o[k] = 0;
        if (threadIdx.x == 0) bias[n] = 0.f;
    }
}

// ------------------------------------------------------- tiled WMMA bf16 GEMM
// C[M,N] = A[M,K] @ Bw[N,K]^T + bias ; EPI==1 -> ReLU.
// Requires M%128==0, N%128==0, K%32==0 (all call sites pad).
// Workgroup tile 128x128, 8 waves (4 along M x 2 along N), wave tile 32x64.
// A staged via async global->LDS (ASYNCcnt), B staged via TDM (TENSORcnt).
template <int EPI>
__global__ __launch_bounds__(256)
void gemm_bf16_wmma(const uint16_t* __restrict__ A, const uint16_t* __restrict__ Bw,
                    const float* __restrict__ bias, float* __restrict__ Cf,
                    uint16_t* __restrict__ Cbf, int M, int N, int K) {
    __shared__ uint16_t sA[2][128][40];   // 40-half stride: conflict-free, 16B aligned
    __shared__ uint16_t sB[2][128][40];
    (void)M;

    const int tid   = threadIdx.x;
    const int wave  = tid >> 5, lane = tid & 31;
    const int waveM = wave >> 1, waveN = wave & 1;
    const int lhalf = lane >> 4, lrow = lane & 15;
    const int rowBase = blockIdx.y * 128;
    const int colBase = blockIdx.x * 128;
    const int lr = tid >> 2;          // 0..63  staging row
    const int lc = (tid & 3) * 8;     // half offset 0/8/16/24

    v8f acc[2][4];
    #pragma unroll
    for (int mt = 0; mt < 2; ++mt)
        #pragma unroll
        for (int nt = 0; nt < 4; ++nt) acc[mt][nt] = {};

    const int nk = K / 32;

    auto stageA = [&](int kk, int buf) {
        const int k0 = kk * 32;
        const uint16_t* g0 = A + (size_t)(rowBase + lr) * K + k0 + lc;
        const uint16_t* g1 = A + (size_t)(rowBase + lr + 64) * K + k0 + lc;
#if USE_ASYNC_LDS
        __builtin_amdgcn_global_load_async_to_lds_b128(
            (__attribute__((address_space(1))) v4i*)g0,
            (__attribute__((address_space(3))) v4i*)&sA[buf][lr][lc], 0, 0);
        __builtin_amdgcn_global_load_async_to_lds_b128(
            (__attribute__((address_space(1))) v4i*)g1,
            (__attribute__((address_space(3))) v4i*)&sA[buf][lr + 64][lc], 0, 0);
#else
        *(uint4*)&sA[buf][lr][lc]      = *(const uint4*)g0;
        *(uint4*)&sA[buf][lr + 64][lc] = *(const uint4*)g1;
#endif
    };
    auto stageB = [&](int kk, int buf) {
        const int k0 = kk * 32;
#if USE_TDM
        if (wave == 0) {
            unsigned ldsOff =
                (unsigned)(size_t)(__attribute__((address_space(3))) uint16_t*)&sB[buf][0][0];
            tdm_load_tile_2d(Bw + (size_t)colBase * K + k0, ldsOff, K, N);
        }
#else
        *(uint4*)&sB[buf][lr][lc] =
            *(const uint4*)(Bw + (size_t)(colBase + lr) * K + k0 + lc);
        *(uint4*)&sB[buf][lr + 64][lc] =
            *(const uint4*)(Bw + (size_t)(colBase + lr + 64) * K + k0 + lc);
#endif
    };
    auto waitStages = [&]() {
#if USE_ASYNC_LDS
        wait_async0();
#endif
#if USE_TDM
        if (wave == 0) wait_tensor0();
#endif
    };

    stageA(0, 0);
    stageB(0, 0);
    waitStages();
    __syncthreads();

    for (int kk = 0; kk < nk; ++kk) {
        const int buf = kk & 1;
        if (kk + 1 < nk) { stageA(kk + 1, buf ^ 1); stageB(kk + 1, buf ^ 1); }
        if (kk + 2 < nk) {  // L2 prefetch of the stage after next -> global_prefetch_b8
            const int k0 = (kk + 2) * 32;
            __builtin_prefetch(A + (size_t)(rowBase + lr) * K + k0 + lc, 0, 1);
            __builtin_prefetch(Bw + (size_t)(colBase + lr) * K + k0 + lc, 0, 1);
        }

        union FragBF { uint4 u[2]; v16bf v; };
        FragBF fa[2], fb[4];
        #pragma unroll
        for (int mt = 0; mt < 2; ++mt) {
            int r = waveM * 32 + mt * 16 + lrow;
            fa[mt].u[0] = *(const uint4*)&sA[buf][r][lhalf * 8];
            fa[mt].u[1] = *(const uint4*)&sA[buf][r][lhalf * 8 + 16];
        }
        #pragma unroll
        for (int nt = 0; nt < 4; ++nt) {
            int r = waveN * 64 + nt * 16 + lrow;
            fb[nt].u[0] = *(const uint4*)&sB[buf][r][lhalf * 8];
            fb[nt].u[1] = *(const uint4*)&sB[buf][r][lhalf * 8 + 16];
        }
        #pragma unroll
        for (int mt = 0; mt < 2; ++mt)
            #pragma unroll
            for (int nt = 0; nt < 4; ++nt)
                acc[mt][nt] = __builtin_amdgcn_wmma_f32_16x16x32_bf16(
                    false, fa[mt].v, false, fb[nt].v, (short)0, acc[mt][nt], false, false);

        if (kk + 1 < nk) waitStages();
        __syncthreads();
    }

    // epilogue: C layout per lane: N = lane&15, M = r + (lane>>4)*8
    #pragma unroll
    for (int nt = 0; nt < 4; ++nt) {
        int cN = colBase + waveN * 64 + nt * 16 + lrow;
        float bv = bias[cN];
        #pragma unroll
        for (int mt = 0; mt < 2; ++mt) {
            #pragma unroll
            for (int r = 0; r < 8; ++r) {
                int row = rowBase + waveM * 32 + mt * 16 + lhalf * 8 + r;
                float v = acc[mt][nt][r] + bv;
                if (EPI == 1) v = fmaxf(v, 0.f);
                if (Cf)  Cf[(size_t)row * N + cN]  = v;
                if (Cbf) Cbf[(size_t)row * N + cN] = f32_to_bf16_rne(v);
            }
        }
    }
}

// --------------------------------------------------------------- gates kernel
__global__ __launch_bounds__(256)
void gates_kernel(const float* __restrict__ preact, const float* __restrict__ c_prev,
                  float* __restrict__ c_t, float* __restrict__ o_t,
                  uint16_t* __restrict__ h_bf) {
    int idx = blockIdx.x * 256 + threadIdx.x;     // over B*H
    int b = idx >> 9, j = idx & 511;
    const float* p = preact + (size_t)b * NPACK;
    float f  = sigmoidf_(p[j]);
    float o  = sigmoidf_(p[j + 512]);
    float i  = sigmoidf_(p[j + 1024]);
    float cn = tanhf(p[j + 1536]);
    float c  = c_prev[idx] * f + i * cn;
    c_t[idx] = c;
    o_t[idx] = o;
    h_bf[idx] = f32_to_bf16_rne(o * tanhf(c));
}

// --------------------------------------------- LCA recall + hpc gate (1 wave/row)
__global__ __launch_bounds__(256)
void lca_kernel(const float* __restrict__ c_t, const float* __restrict__ o_t,
                const float* __restrict__ dec1, const float* __restrict__ em,
                const float* __restrict__ Whpc, const float* __restrict__ bhpc,
                float* __restrict__ cm_out, float* __restrict__ h2_out,
                uint16_t* __restrict__ h2_bf) {
    int wave = threadIdx.x >> 5, lane = threadIdx.x & 31;
    int b = blockIdx.x * 8 + wave;
    const float* c  = c_t  + (size_t)b * H_SZ;
    const float* e0 = em   + (size_t)b * H_SZ;
    const float* e1 = em   + (size_t)B_SZ * H_SZ + (size_t)b * H_SZ;
    const float* d  = dec1 + (size_t)b * H_SZ;
    const int base = lane * 16;

    float cc[16], a0[16], a1[16];
    float cn2 = 0, n0 = 0, n1 = 0, d0 = 0, d1 = 0, hc = 0, hd = 0, w0 = 0, w1 = 0;
    #pragma unroll
    for (int i = 0; i < 16; ++i) {
        float vc = c[base + i], v0 = e0[base + i], v1 = e1[base + i];
        cc[i] = vc; a0[i] = v0; a1[i] = v1;
        cn2 += vc * vc;  n0 += v0 * v0;  n1 += v1 * v1;
        d0  += vc * v0;  d1 += vc * v1;
        hc  += Whpc[512 + base + i]  * vc;
        hd  += Whpc[1024 + base + i] * d[base + i];
        w0  += Whpc[base + i] * v0;
        w1  += Whpc[base + i] * v1;
    }
    #pragma unroll
    for (int off = 16; off > 0; off >>= 1) {
        cn2 += __shfl_xor(cn2, off, 32);  n0 += __shfl_xor(n0, off, 32);
        n1  += __shfl_xor(n1, off, 32);   d0 += __shfl_xor(d0, off, 32);
        d1  += __shfl_xor(d1, off, 32);   hc += __shfl_xor(hc, off, 32);
        hd  += __shfl_xor(hd, off, 32);   w0 += __shfl_xor(w0, off, 32);
        w1  += __shfl_xor(w1, off, 32);
    }
    float an  = fmaxf(sqrtf(cn2), 1e-8f);
    float bn0 = fmaxf(sqrtf(n0), 1e-8f), bn1 = fmaxf(sqrtf(n1), 1e-8f);
    float s0 = d0 / (an * bn0), s1 = d1 / (an * bn1);
    float x0 = 0.f, x1 = 0.f;
    #pragma unroll
    for (int it = 0; it < 10; ++it) {          // LCA dynamics, K=2: inhib = other's relu
        float f0 = fmaxf(x0, 0.f), f1 = fmaxf(x1, 0.f);
        x0 = fmaxf(x0 + 0.5f * (0.3f * s0 - 0.8f * f1), 0.f);
        x1 = fmaxf(x1 + 0.5f * (0.3f * s1 - 0.8f * f0), 0.f);
    }
    float g = sigmoidf_(x0 * w0 + x1 * w1 + hc + hd + bhpc[0]);
    #pragma unroll
    for (int i = 0; i < 16; ++i) {
        float m  = x0 * a0[i] + x1 * a1[i];
        float cm = cc[i] + m * g;
        size_t gi = (size_t)b * H_SZ + base + i;
        cm_out[gi] = cm;
        float h2 = o_t[gi] * tanhf(cm);
        h2_out[gi] = h2;
        h2_bf[gi]  = f32_to_bf16_rne(h2);
    }
}

// ---------------------------------------------------- softmax + value (1 wave/row)
__global__ __launch_bounds__(256)
void softmax_kernel(const float* __restrict__ logits, float* __restrict__ pi,
                    float* __restrict__ value) {
    int wave = threadIdx.x >> 5, lane = threadIdx.x & 31;
    int b = blockIdx.x * 8 + wave;
    const float* r = logits + (size_t)b * NAC;
    float v0 = r[lane], v1 = r[lane + 32];
    float m = fmaxf(v0, v1);
    #pragma unroll
    for (int off = 16; off > 0; off >>= 1) m = fmaxf(m, __shfl_xor(m, off, 32));
    float e0 = __expf(v0 - m), e1 = __expf(v1 - m);
    float s = e0 + e1;
    #pragma unroll
    for (int off = 16; off > 0; off >>= 1) s += __shfl_xor(s, off, 32);
    float inv = 1.0f / s;
    pi[(size_t)b * 64 + lane]      = e0 * inv;
    pi[(size_t)b * 64 + lane + 32] = e1 * inv;
    if (lane == 0) value[b] = r[64];
}

// --------------------------------------------------------------------- launch
extern "C" void kernel_launch(void* const* d_in, const int* in_sizes, int n_in,
                              void* d_out, int out_size, void* d_ws, size_t ws_size,
                              hipStream_t stream) {
    const float* x    = (const float*)d_in[0];
    const float* h    = (const float*)d_in[1];
    const float* cp   = (const float*)d_in[2];
    const float* em   = (const float*)d_in[3];
    const float* Wi   = (const float*)d_in[4];
    const float* bi   = (const float*)d_in[5];
    const float* Wh   = (const float*)d_in[6];
    const float* bh   = (const float*)d_in[7];
    const float* Wih  = (const float*)d_in[8];
    const float* bih  = (const float*)d_in[9];
    const float* Wa   = (const float*)d_in[10];
    const float* ba   = (const float*)d_in[11];
    const float* Wc   = (const float*)d_in[12];
    const float* bc   = (const float*)d_in[13];
    const float* Whpc = (const float*)d_in[14];
    const float* bhpc = (const float*)d_in[15];

    // workspace carve-up (256B aligned)
    char* ws = (char*)d_ws;
    size_t off = 0;
    auto alloc = [&](size_t bytes) -> void* {
        void* p = ws + off;
        off = (off + bytes + 255) & ~(size_t)255;
        return p;
    };
    uint16_t* actp    = (uint16_t*)alloc((size_t)B_SZ * KPACK * 2);
    uint16_t* Wp1     = (uint16_t*)alloc((size_t)NPACK * KPACK * 2);
    float*    bias1   = (float*)   alloc((size_t)NPACK * 4);
    float*    preact  = (float*)   alloc((size_t)B_SZ * NPACK * 4);
    float*    c_t     = (float*)   alloc((size_t)B_SZ * H_SZ * 4);
    float*    o_t     = (float*)   alloc((size_t)B_SZ * H_SZ * 4);
    uint16_t* h_bf    = (uint16_t*)alloc((size_t)B_SZ * H_SZ * 2);
    uint16_t* Wih_bf  = (uint16_t*)alloc((size_t)DECSZ * H_SZ * 2);
    float*    dec1    = (float*)   alloc((size_t)B_SZ * DECSZ * 4);
    uint16_t* h2_bf   = (uint16_t*)alloc((size_t)B_SZ * H_SZ * 2);
    uint16_t* dec2_bf = (uint16_t*)alloc((size_t)B_SZ * DECSZ * 2);
    uint16_t* Wac_bf  = (uint16_t*)alloc((size_t)NAC * DECSZ * 2);
    float*    bias_ac = (float*)   alloc((size_t)NAC * 4);
    float*    logits  = (float*)   alloc((size_t)B_SZ * NAC * 4);

    // output carve-up: pi(B*64) | value(B) | h_t2(B*512) | cm_t(B*512)
    float* out    = (float*)d_out;
    float* pi     = out;
    float* value  = out + (size_t)B_SZ * 64;
    float* h2_out = value + B_SZ;
    float* cm_out = h2_out + (size_t)B_SZ * H_SZ;

    // 1) packing / bf16 conversion
    pack_act_kernel<<<B_SZ, 256, 0, stream>>>(x, h, actp);
    pack_w1_kernel<<<NPACK, 256, 0, stream>>>(Wi, bi, Wh, bh, Wp1, bias1);
    cvt_bf16_kernel<<<(DECSZ * H_SZ + 255) / 256, 256, 0, stream>>>(Wih, Wih_bf, DECSZ * H_SZ);
    pack_wac_kernel<<<NAC, 256, 0, stream>>>(Wa, ba, Wc, bc, Wac_bf, bias_ac);

    // 2) GEMM1: preact = [x|h] @ [Wi|Wh]^T + (bi+bh)   (51.6 GFLOP, WMMA bf16)
    gemm_bf16_wmma<0><<<dim3(NPACK / 128, B_SZ / 128), 256, 0, stream>>>(
        actp, Wp1, bias1, preact, (uint16_t*)nullptr, B_SZ, NPACK, KPACK);

    // 3) gates -> c_t, o_t, h_t(bf16)
    gates_kernel<<<(B_SZ * H_SZ) / 256, 256, 0, stream>>>(preact, cp, c_t, o_t, h_bf);

    // 4) GEMM2: dec1 = relu(h_t @ W_ih^T + b_ih)
    gemm_bf16_wmma<1><<<dim3(DECSZ / 128, B_SZ / 128), 256, 0, stream>>>(
        h_bf, Wih_bf, bih, dec1, (uint16_t*)nullptr, B_SZ, DECSZ, H_SZ);

    // 5) LCA recall + hpc gate -> cm_t (out), h_t2 (out + bf16)
    lca_kernel<<<B_SZ / 8, 256, 0, stream>>>(c_t, o_t, dec1, em, Whpc, bhpc,
                                             cm_out, h2_out, h2_bf);

    // 6) GEMM3: dec2 = relu(h_t2 @ W_ih^T + b_ih)  (bf16 out for GEMM4)
    gemm_bf16_wmma<1><<<dim3(DECSZ / 128, B_SZ / 128), 256, 0, stream>>>(
        h2_bf, Wih_bf, bih, (float*)nullptr, dec2_bf, B_SZ, DECSZ, H_SZ);

    // 7) GEMM4: [actor|critic|pad] logits, N padded to 128
    gemm_bf16_wmma<0><<<dim3(NAC / 128, B_SZ / 128), 256, 0, stream>>>(
        dec2_bf, Wac_bf, bias_ac, logits, (uint16_t*)nullptr, B_SZ, NAC, DECSZ);

    // 8) softmax over 64 actions + value write
    softmax_kernel<<<B_SZ / 8, 256, 0, stream>>>(logits, pi, value);

    (void)in_sizes; (void)n_in; (void)out_size; (void)ws_size;
}
